// MoEAD_43611097924200
// MI455X (gfx1250) — matrix operand order repaired and checked by reading, not compiled, not run on hardware
//
#include <hip/hip_runtime.h>
#include <hip/hip_bf16.h>

typedef _Float16 half_t;
typedef __attribute__((ext_vector_type(16))) _Float16 v16h;
typedef __attribute__((ext_vector_type(8)))  _Float16 v8h;
typedef __attribute__((ext_vector_type(8)))  float    v8f;

#define C_B    4
#define C_CIN  272
#define C_HW   1024
#define C_D    256
#define C_NH   8
#define C_DH   32
#define C_DFF  1024
#define C_E    8
#define C_N    4096      // HW * B tokens
#define C_KPAD 288       // CIN padded to multiple of 32

// ---------------- WMMA helpers ----------------

__device__ inline v8f wmma_f16(v16h a, v16h b, v8f c) {
  return __builtin_amdgcn_wmma_f32_16x16x32_f16(false, a, false, b, (short)0, c, false, false);
}

// A-matrix fragment (16xK tile, K=32 slice). Row = lane&15.
// lanes 0-15: K 0..7 (v0-3), 16..23 (v4-7); lanes 16-31: K 8..15, 24..31.
__device__ inline v16h frag_ld_a(const half_t* p0, int ld) {
  const int lane = threadIdx.x & 31;
  const half_t* p = p0 + (size_t)(lane & 15) * ld + ((lane >> 4) << 3);
  v16h f;
  *reinterpret_cast<v8h*>(&f)       = *reinterpret_cast<const v8h*>(p);
  *(reinterpret_cast<v8h*>(&f) + 1) = *reinterpret_cast<const v8h*>(p + 16);
  return f;
}

// B-operand fragment loaded from B^T rows ([N][K] row-major, e.g. weight [out][in]).
// lanes 0-15: K 0..7, 16..23; lanes 16-31: K 8..15, 24..31. Column N = lane&15.
__device__ inline v16h frag_ld_b(const half_t* p0, int ld) {
  const int lane = threadIdx.x & 31;
  const half_t* p = p0 + (size_t)(lane & 15) * ld + ((lane >> 4) << 3);
  v16h f;
  *reinterpret_cast<v8h*>(&f)       = *reinterpret_cast<const v8h*>(p);
  *(reinterpret_cast<v8h*>(&f) + 1) = *reinterpret_cast<const v8h*>(p + 16);
  return f;
}

__device__ inline float hmax16(float v) {
#pragma unroll
  for (int m = 1; m < 16; m <<= 1) v = fmaxf(v, __shfl_xor(v, m, 32));
  return v;
}
__device__ inline float hsum16(float v) {
#pragma unroll
  for (int m = 1; m < 16; m <<= 1) v += __shfl_xor(v, m, 32);
  return v;
}

// ---------------- Generic WMMA GEMM ----------------
// Y[N,M] = act(A[N,K] @ W[M,K]^T + bias) * rowscale, optionally += into Yf.
__global__ void k_gemm(const half_t* __restrict__ A, int lda,
                       const half_t* __restrict__ W,
                       const float* __restrict__ bias,
                       float* __restrict__ Yf, half_t* __restrict__ Yh,
                       const float* __restrict__ rowscale, int rs_stride,
                       int Nrows, int K, int M, int relu, int accum) {
  const int colTiles = M >> 4;
  const int tiles = (Nrows >> 4) * colTiles;
  const int wave = blockIdx.x * (blockDim.x >> 5) + (threadIdx.x >> 5);
  if (wave >= tiles) return;                 // wave-uniform; EXEC stays all-ones
  const int rt = wave / colTiles, ct = wave % colTiles;
  const half_t* Ab = A + (size_t)(rt << 4) * lda;
  const half_t* Wb = W + (size_t)(ct << 4) * K;
  v8f acc = {0.f, 0.f, 0.f, 0.f, 0.f, 0.f, 0.f, 0.f};
  for (int kk = 0; kk < K; kk += 32) {
    v16h a = frag_ld_a(Ab + kk, lda);
    v16h b = frag_ld_b(Wb + kk, K);
    acc = wmma_f16(a, b, acc);
  }
  const int lane = threadIdx.x & 31;
  const int col = (ct << 4) + (lane & 15);
  const float bv = bias ? bias[col] : 0.f;
#pragma unroll
  for (int j = 0; j < 8; ++j) {
    const int row = (rt << 4) + ((lane < 16) ? j : j + 8);
    float v = acc[j] + bv;
    if (relu) v = fmaxf(v, 0.f);
    if (rowscale) v *= rowscale[(size_t)row * rs_stride];
    const size_t idx = (size_t)row * M + col;
    if (Yf) { if (accum) Yf[idx] += v; else Yf[idx] = v; }
    if (Yh) Yh[idx] = (half_t)v;
  }
}

// ---------------- Fused flash attention ----------------
// One wave per (b, h, 16-query tile). Q/K in [N=4096][256] f16 (head cols h*32..),
// Vt in [b][h][dh=32][S=1024] f16. Output f16 [N][256].
__global__ void k_attn(const half_t* __restrict__ Q, const half_t* __restrict__ Km,
                       const half_t* __restrict__ Vt, half_t* __restrict__ O) {
  __shared__ __align__(16) half_t plds[8][16][32];
  const int wv = threadIdx.x >> 5;
  const int wave = blockIdx.x * 8 + wv;          // 0..2047
  const int st = wave & 63;
  const int bh = wave >> 6;
  const int b = bh & 3, h = bh >> 2;
  const int lane = threadIdx.x & 31;

  const half_t* Qbase = Q + ((size_t)(st * 16) * 4 + b) * 256 + h * 32;
  v16h qa = frag_ld_a(Qbase, 1024);              // 16 queries x 32 (full head dim)

  float mrow[8], lrow[8];
  v8f o0 = {0.f,0.f,0.f,0.f,0.f,0.f,0.f,0.f};
  v8f o1 = {0.f,0.f,0.f,0.f,0.f,0.f,0.f,0.f};
#pragma unroll
  for (int j = 0; j < 8; ++j) { mrow[j] = -1e30f; lrow[j] = 0.f; }
  const float sc = 0.17677669529663687f;         // 1/sqrt(32)
  const half_t* Vb = Vt + (size_t)((b * 8 + h) * 32) * 1024;

  for (int kt = 0; kt < 1024; kt += 32) {
    v16h kb0 = frag_ld_b(Km + ((size_t)(kt) * 4 + b) * 256 + h * 32, 1024);
    v16h kb1 = frag_ld_b(Km + ((size_t)(kt + 16) * 4 + b) * 256 + h * 32, 1024);
    v8f z0 = {0.f,0.f,0.f,0.f,0.f,0.f,0.f,0.f};
    v8f s0 = wmma_f16(qa, kb0, z0);
    v8f s1 = wmma_f16(qa, kb1, z0);
#pragma unroll
    for (int j = 0; j < 8; ++j) {
      float a0 = s0[j] * sc, a1 = s1[j] * sc;
      float tm = hmax16(fmaxf(a0, a1));
      float nm = fmaxf(mrow[j], tm);
      float p0 = __expf(a0 - nm), p1 = __expf(a1 - nm);
      float rs = hsum16(p0 + p1);
      float alpha = __expf(mrow[j] - nm);
      lrow[j] = lrow[j] * alpha + rs;
      mrow[j] = nm;
      o0[j] *= alpha; o1[j] *= alpha;
      const int r = (lane < 16) ? j : j + 8;
      const int c = lane & 15;
      plds[wv][r][c]      = (half_t)p0;
      plds[wv][r][c + 16] = (half_t)p1;
    }
    __syncthreads();
    v16h pa  = frag_ld_a(&plds[wv][0][0], 32);
    v16h vb0 = frag_ld_b(Vb + kt, 1024);
    v16h vb1 = frag_ld_b(Vb + (size_t)16 * 1024 + kt, 1024);
    o0 = wmma_f16(pa, vb0, o0);
    o1 = wmma_f16(pa, vb1, o1);
    __syncthreads();
  }
#pragma unroll
  for (int j = 0; j < 8; ++j) {
    const int r = (lane < 16) ? j : j + 8;
    const size_t n = (size_t)(st * 16 + r) * 4 + b;
    const int c = lane & 15;
    const float inv = 1.0f / lrow[j];
    O[n * 256 + h * 32 + c]      = (half_t)(o0[j] * inv);
    O[n * 256 + h * 32 + 16 + c] = (half_t)(o1[j] * inv);
  }
}

// ---------------- LayerNorm (+ optional residual), dual output ----------------
__global__ void k_ln(const float* __restrict__ A, const float* __restrict__ R,
                     const float* __restrict__ g, const float* __restrict__ be,
                     float* __restrict__ Of, half_t* __restrict__ Oh) {
  __shared__ float sm[256];
  const int row = blockIdx.x, t = threadIdx.x;
  float v = A[(size_t)row * 256 + t];
  if (R) v += R[(size_t)row * 256 + t];
  sm[t] = v; __syncthreads();
  for (int s = 128; s > 0; s >>= 1) { if (t < s) sm[t] += sm[t + s]; __syncthreads(); }
  const float mean = sm[0] * (1.0f / 256.0f);
  __syncthreads();
  const float d = v - mean;
  sm[t] = d * d; __syncthreads();
  for (int s = 128; s > 0; s >>= 1) { if (t < s) sm[t] += sm[t + s]; __syncthreads(); }
  const float var = sm[0] * (1.0f / 256.0f);
  const float o = d * rsqrtf(var + 1e-5f) * g[t] + be[t];
  if (Of) Of[(size_t)row * 256 + t] = o;
  if (Oh) Oh[(size_t)row * 256 + t] = (half_t)o;
}

// ---------------- Small elementwise kernels ----------------

__global__ void k_pos(float* __restrict__ pos) {
  const int i = blockIdx.x * blockDim.x + threadIdx.x;
  if (i >= C_HW * C_D) return;
  const int s = i >> 8, d = i & 255;
  const int h = s >> 5, w = s & 31;
  const float scale = 6.283185307179586f;
  float coord; int dd;
  if (d < 128) { coord = (float)(h + 1) / (32.0f + 1e-6f) * scale; dd = d; }
  else         { coord = (float)(w + 1) / (32.0f + 1e-6f) * scale; dd = d - 128; }
  const int ii = dd >> 1;
  const float dt = powf(10000.0f, (float)(2 * ii) / 128.0f);
  const float a = coord / dt;
  pos[i] = (dd & 1) ? cosf(a) : sinf(a);
}

__global__ void k_tokens(const float* __restrict__ x, half_t* __restrict__ X) {
  const int i = blockIdx.x * blockDim.x + threadIdx.x;
  if (i >= C_N * C_KPAD) return;
  const int n = i / C_KPAD, c = i % C_KPAD;
  const int s = n >> 2, b = n & 3;
  X[i] = (half_t)((c < C_CIN) ? x[(size_t)(b * C_CIN + c) * C_HW + s] : 0.f);
}

__global__ void k_cvt_pad(const float* __restrict__ S, half_t* __restrict__ D_,
                          int rows, int K, int Kp) {
  const int i = blockIdx.x * blockDim.x + threadIdx.x;
  if (i >= rows * Kp) return;
  const int r = i / Kp, k = i % Kp;
  D_[i] = (half_t)((k < K) ? S[(size_t)r * K + k] : 0.f);
}

// [e][r][c] f32 -> [e][c][r] f16
__global__ void k_cvt_t(const float* __restrict__ S, half_t* __restrict__ D_,
                        int E_, int R, int C) {
  const int i = blockIdx.x * blockDim.x + threadIdx.x;
  if (i >= E_ * R * C) return;
  const int e = i / (R * C), rem = i % (R * C);
  const int r = rem / C, c = rem % C;
  D_[(size_t)e * R * C + (size_t)c * R + r] = (half_t)S[i];
}

// out = (half)(X + pos[token_s])   (X is [N=4096][256], token s = row/4)
__global__ void k_add_pos(const float* __restrict__ X, const float* __restrict__ pos,
                          half_t* __restrict__ O) {
  const int i = blockIdx.x * blockDim.x + threadIdx.x;
  if (i >= C_N * C_D) return;
  const int d = i & 255;
  O[i] = (half_t)(X[i] + pos[(size_t)(i >> 10) * 256 + d]);
}

// tgt init: tf = learned (broadcast over batch); qh = (half)(learned + pos)
__global__ void k_bcast(const float* __restrict__ lem, const float* __restrict__ pos,
                        float* __restrict__ tf, half_t* __restrict__ qh) {
  const int i = blockIdx.x * blockDim.x + threadIdx.x;
  if (i >= C_N * C_D) return;
  const int d = i & 255;
  const size_t si = (size_t)(i >> 10) * 256 + d;
  const float l = lem[si];
  tf[i] = l;
  qh[i] = (half_t)(l + pos[si]);
}

// V [N=4096][256] f16 -> Vt [b][h][dh=32][S=1024] f16
__global__ void k_tv(const half_t* __restrict__ V, half_t* __restrict__ Vt) {
  const int i = blockIdx.x * blockDim.x + threadIdx.x;
  if (i >= C_N * C_D) return;
  const int n = i >> 8, d = i & 255;
  const int b = n & 3, s = n >> 2, h = d >> 5, dd = d & 31;
  Vt[(size_t)((b * 8 + h) * 32 + dd) * 1024 + s] = V[i];
}

__global__ void k_gate(const float* __restrict__ X, const float* __restrict__ Wg,
                       const float* __restrict__ bg, float* __restrict__ L) {
  const int i = blockIdx.x * blockDim.x + threadIdx.x;
  if (i >= C_N * C_E) return;
  const int n = i >> 3, e = i & 7;
  float s = bg[e];
  const float* xp = X + (size_t)n * 256;
  const float* wp = Wg + (size_t)e * 256;
  for (int d = 0; d < 256; ++d) s += xp[d] * wp[d];
  L[i] = s;
}

__global__ void k_top2(const float* __restrict__ L, float* __restrict__ C) {
  const int n = blockIdx.x * blockDim.x + threadIdx.x;
  if (n >= C_N) return;
  float v[C_E];
#pragma unroll
  for (int e = 0; e < C_E; ++e) v[e] = L[(size_t)n * C_E + e];
  int i1 = 0;
#pragma unroll
  for (int e = 1; e < C_E; ++e) if (v[e] > v[i1]) i1 = e;   // first max on tie
  int i2 = -1;
#pragma unroll
  for (int e = 0; e < C_E; ++e) if (e != i1 && (i2 < 0 || v[e] > v[i2])) i2 = e;
  const float e2 = __expf(v[i2] - v[i1]);
  const float inv = 1.0f / (1.0f + e2);
#pragma unroll
  for (int e = 0; e < C_E; ++e) C[(size_t)n * C_E + e] = 0.f;
  C[(size_t)n * C_E + i1] = inv;
  C[(size_t)n * C_E + i2] = e2 * inv;
}

__global__ void k_zero(float* __restrict__ p, int n) {
  const int i = blockIdx.x * blockDim.x + threadIdx.x;
  if (i < n) p[i] = 0.f;
}

// rec [N=4096][272] f32 -> out [B][CIN][H][W]
__global__ void k_out(const float* __restrict__ rec, float* __restrict__ out) {
  const int i = blockIdx.x * blockDim.x + threadIdx.x;
  if (i >= C_B * C_CIN * C_HW) return;
  const int s = i & 1023;
  const int bc = i >> 10;
  const int c = bc % C_CIN, b = bc / C_CIN;
  out[i] = rec[(size_t)(s * 4 + b) * C_CIN + c];
}

// ---------------- Host orchestration ----------------

extern "C" void kernel_launch(void* const* d_in, const int* in_sizes, int n_in,
                              void* d_out, int out_size, void* d_ws, size_t ws_size,
                              hipStream_t stream) {
  (void)in_sizes; (void)n_in; (void)out_size; (void)ws_size;
  const float* x    = (const float*)d_in[0];
  const float* Wi   = (const float*)d_in[1];
  const float* bi   = (const float*)d_in[2];
  const float* lem  = (const float*)d_in[3];
  const float* Wqs  = (const float*)d_in[4];
  const float* bqs  = (const float*)d_in[5];
  const float* Wos  = (const float*)d_in[6];
  const float* bos  = (const float*)d_in[7];
  const float* Wqc  = (const float*)d_in[8];
  const float* bqc  = (const float*)d_in[9];
  const float* Woc  = (const float*)d_in[10];
  const float* boc  = (const float*)d_in[11];
  const float* lng  = (const float*)d_in[12];
  const float* lnb  = (const float*)d_in[13];
  const float* gw1  = (const float*)d_in[14];
  const float* gb1  = (const float*)d_in[15];
  const float* W1a  = (const float*)d_in[16];
  const float* b1a  = (const float*)d_in[17];
  const float* W2a  = (const float*)d_in[18];
  const float* b2a  = (const float*)d_in[19];
  const float* gw2  = (const float*)d_in[20];
  const float* gb2  = (const float*)d_in[21];
  const float* W1b  = (const float*)d_in[22];
  const float* b1b  = (const float*)d_in[23];
  const float* W2b  = (const float*)d_in[24];
  const float* b2b  = (const float*)d_in[25];
  const float* Wout = (const float*)d_in[26];
  const float* bout = (const float*)d_in[27];
  float* outp = (float*)d_out;

  char* base = (char*)d_ws;
  size_t off = 0;
  auto alloc = [&](size_t bytes) -> void* {
    void* p = base + off;
    off += (bytes + 255) & ~(size_t)255;
    return p;
  };
  float*  POS   = (float*) alloc((size_t)C_HW * C_D * 4);
  half_t* XT    = (half_t*)alloc((size_t)C_N * C_KPAD * 2);
  half_t* WIh   = (half_t*)alloc((size_t)C_D * C_KPAD * 2);
  half_t* WQSh  = (half_t*)alloc((size_t)3 * C_D * C_D * 2);
  half_t* WOSh  = (half_t*)alloc((size_t)C_D * C_D * 2);
  half_t* WQCh  = (half_t*)alloc((size_t)3 * C_D * C_D * 2);
  half_t* WOCh  = (half_t*)alloc((size_t)C_D * C_D * 2);
  half_t* W1AT  = (half_t*)alloc((size_t)C_E * C_D * C_DFF * 2);
  half_t* W2AT  = (half_t*)alloc((size_t)C_E * C_DFF * C_D * 2);
  half_t* W1BT  = (half_t*)alloc((size_t)C_E * C_D * C_DFF * 2);
  half_t* W2BT  = (half_t*)alloc((size_t)C_E * C_DFF * C_D * 2);
  half_t* WOUTh = (half_t*)alloc((size_t)C_CIN * C_D * 2);
  float*  SRCF  = (float*) alloc((size_t)C_N * C_D * 4);
  half_t* SRCH  = (half_t*)alloc((size_t)C_N * C_D * 2);
  half_t* ENCPH = (half_t*)alloc((size_t)C_N * C_D * 2);
  float*  TGTF  = (float*) alloc((size_t)C_N * C_D * 4);
  half_t* TGTH  = (half_t*)alloc((size_t)C_N * C_D * 2);
  half_t* QINH  = (half_t*)alloc((size_t)C_N * C_D * 2);
  half_t* QH    = (half_t*)alloc((size_t)C_N * C_D * 2);
  half_t* KH    = (half_t*)alloc((size_t)C_N * C_D * 2);
  half_t* VH    = (half_t*)alloc((size_t)C_N * C_D * 2);
  half_t* VTH   = (half_t*)alloc((size_t)C_N * C_D * 2);
  half_t* AOH   = (half_t*)alloc((size_t)C_N * C_D * 2);
  float*  T2F   = (float*) alloc((size_t)C_N * C_D * 4);
  float*  LOG   = (float*) alloc((size_t)C_N * C_E * 4);
  float*  COMB  = (float*) alloc((size_t)C_N * C_E * 4);
  half_t* HEH   = (half_t*)alloc((size_t)C_N * C_DFF * 2);
  float*  MOEF  = (float*) alloc((size_t)C_N * C_D * 4);
  float*  ODF   = (float*) alloc((size_t)C_N * C_D * 4);
  half_t* ODH   = (half_t*)alloc((size_t)C_N * C_D * 2);
  float*  DECF  = (float*) alloc((size_t)C_N * C_D * 4);
  half_t* DECH  = (half_t*)alloc((size_t)C_N * C_D * 2);
  half_t* DECPH = (half_t*)alloc((size_t)C_N * C_D * 2);
  float*  RECF  = (float*) alloc((size_t)C_N * C_CIN * 4);

  auto grid1 = [](int n) { return dim3((unsigned)((n + 255) / 256)); };
  auto gemm = [&](const half_t* A, int lda, const half_t* W, const float* bias,
                  float* Yf, half_t* Yh, const float* rs, int rss,
                  int N_, int K_, int M_, int relu, int acc) {
    const int tiles = (N_ / 16) * (M_ / 16);
    k_gemm<<<dim3((unsigned)((tiles + 7) / 8)), dim3(256), 0, stream>>>(
        A, lda, W, bias, Yf, Yh, rs, rss, N_, K_, M_, relu, acc);
  };
  auto ln = [&](const float* A, const float* R, int li, float* Of, half_t* Oh) {
    k_ln<<<dim3(C_N), dim3(256), 0, stream>>>(A, R, lng + (size_t)li * C_D,
                                              lnb + (size_t)li * C_D, Of, Oh);
  };

  // ---- prep ----
  k_pos   <<<grid1(C_HW * C_D), 256, 0, stream>>>(POS);
  k_tokens<<<grid1(C_N * C_KPAD), 256, 0, stream>>>(x, XT);
  k_cvt_pad<<<grid1(C_D * C_KPAD), 256, 0, stream>>>(Wi, WIh, C_D, C_CIN, C_KPAD);
  k_cvt_pad<<<grid1(3 * C_D * C_D), 256, 0, stream>>>(Wqs, WQSh, 3 * C_D, C_D, C_D);
  k_cvt_pad<<<grid1(C_D * C_D), 256, 0, stream>>>(Wos, WOSh, C_D, C_D, C_D);
  k_cvt_pad<<<grid1(3 * C_D * C_D), 256, 0, stream>>>(Wqc, WQCh, 3 * C_D, C_D, C_D);
  k_cvt_pad<<<grid1(C_D * C_D), 256, 0, stream>>>(Woc, WOCh, C_D, C_D, C_D);
  k_cvt_pad<<<grid1(C_CIN * C_D), 256, 0, stream>>>(Wout, WOUTh, C_CIN, C_D, C_D);
  k_cvt_t<<<grid1(C_E * C_D * C_DFF), 256, 0, stream>>>(W1a, W1AT, C_E, C_D, C_DFF);
  k_cvt_t<<<grid1(C_E * C_DFF * C_D), 256, 0, stream>>>(W2a, W2AT, C_E, C_DFF, C_D);
  k_cvt_t<<<grid1(C_E * C_D * C_DFF), 256, 0, stream>>>(W1b, W1BT, C_E, C_D, C_DFF);
  k_cvt_t<<<grid1(C_E * C_DFF * C_D), 256, 0, stream>>>(W2b, W2BT, C_E, C_DFF, C_D);

  // input projection: src = tokens @ Wi^T + bi
  gemm(XT, C_KPAD, WIh, bi, SRCF, SRCH, nullptr, 0, C_N, C_KPAD, C_D, 0, 0);
  k_add_pos<<<grid1(C_N * C_D), 256, 0, stream>>>(SRCF, POS, ENCPH);

  for (int i = 0; i < 2; ++i) {
    const float*  decF  = i ? DECF : SRCF;
    const half_t* decH  = i ? DECH : SRCH;
    const half_t* decPH = i ? DECPH : ENCPH;

    // tgt = learned_embed (broadcast); q_in = tgt + pos
    k_bcast<<<grid1(C_N * C_D), 256, 0, stream>>>(lem, POS, TGTF, QINH);

    // ---- self attention (K/V from enc = src) ----
    gemm(QINH, C_D, WQSh,                 bqs,         nullptr, QH, nullptr, 0, C_N, C_D, C_D, 0, 0);
    gemm(ENCPH, C_D, WQSh + 256 * 256,    bqs + 256,   nullptr, KH, nullptr, 0, C_N, C_D, C_D, 0, 0);
    gemm(SRCH, C_D, WQSh + 512 * 256,     bqs + 512,   nullptr, VH, nullptr, 0, C_N, C_D, C_D, 0, 0);
    k_tv<<<grid1(C_N * C_D), 256, 0, stream>>>(VH, VTH);
    k_attn<<<dim3(256), dim3(256), 0, stream>>>(QH, KH, VTH, AOH);
    gemm(AOH, C_D, WOSh, bos, T2F, nullptr, nullptr, 0, C_N, C_D, C_D, 0, 0);
    ln(TGTF, T2F, 4 * i + 0, TGTF, TGTH);

    // ---- cross attention (K/V from dec) ----
    k_add_pos<<<grid1(C_N * C_D), 256, 0, stream>>>(TGTF, POS, QINH);
    gemm(QINH, C_D, WQCh,                 bqc,         nullptr, QH, nullptr, 0, C_N, C_D, C_D, 0, 0);
    gemm(decPH, C_D, WQCh + 256 * 256,    bqc + 256,   nullptr, KH, nullptr, 0, C_N, C_D, C_D, 0, 0);
    gemm(decH, C_D, WQCh + 512 * 256,     bqc + 512,   nullptr, VH, nullptr, 0, C_N, C_D, C_D, 0, 0);
    k_tv<<<grid1(C_N * C_D), 256, 0, stream>>>(VH, VTH);
    k_attn<<<dim3(256), dim3(256), 0, stream>>>(QH, KH, VTH, AOH);
    gemm(AOH, C_D, WOCh, boc, T2F, nullptr, nullptr, 0, C_N, C_D, C_D, 0, 0);
    ln(TGTF, T2F, 4 * i + 1, TGTF, TGTH);

    // ---- MoE 1 ----
    k_gate<<<grid1(C_N * C_E), 256, 0, stream>>>(TGTF, gw1, gb1, LOG);
    k_top2<<<grid1(C_N), 256, 0, stream>>>(LOG, COMB);
    k_zero<<<grid1(C_N * C_D), 256, 0, stream>>>(MOEF, C_N * C_D);
    for (int e = 0; e < C_E; ++e) {
      gemm(TGTH, C_D, W1AT + (size_t)e * C_DFF * C_D, b1a + (size_t)e * C_DFF,
           nullptr, HEH, nullptr, 0, C_N, C_D, C_DFF, 1, 0);
      gemm(HEH, C_DFF, W2AT + (size_t)e * C_D * C_DFF, b2a + (size_t)e * C_D,
           MOEF, nullptr, COMB + e, C_E, C_N, C_DFF, C_D, 0, 1);
    }
    ln(TGTF, MOEF, 4 * i + 2, ODF, ODH);

    // ---- MoE 2 ----
    k_gate<<<grid1(C_N * C_E), 256, 0, stream>>>(ODF, gw2, gb2, LOG);
    k_top2<<<grid1(C_N), 256, 0, stream>>>(LOG, COMB);
    k_zero<<<grid1(C_N * C_D), 256, 0, stream>>>(MOEF, C_N * C_D);
    for (int e = 0; e < C_E; ++e) {
      gemm(ODH, C_D, W1BT + (size_t)e * C_DFF * C_D, b1b + (size_t)e * C_DFF,
           nullptr, HEH, nullptr, 0, C_N, C_D, C_DFF, 1, 0);
      gemm(HEH, C_DFF, W2BT + (size_t)e * C_D * C_DFF, b2b + (size_t)e * C_D,
           MOEF, nullptr, COMB + e, C_E, C_N, C_DFF, C_D, 0, 1);
    }
    ln(ODF, MOEF, 4 * i + 3, DECF, DECH);
    k_add_pos<<<grid1(C_N * C_D), 256, 0, stream>>>(DECF, POS, DECPH);
  }

  // output projection + scatter to [B, CIN, H, W]
  gemm(DECH, C_D, WOUTh, bout, RECF, nullptr, nullptr, 0, C_N, C_D, C_CIN, 0, 0);
  k_out<<<grid1(C_B * C_CIN * C_HW), 256, 0, stream>>>(RECF, outp);
}
